// MultiheadAttention_82403242541592
// MI455X (gfx1250) — matrix operand order repaired and checked by reading, not compiled, not run on hardware
//
#include <hip/hip_runtime.h>
#include <hip/hip_bf16.h>

// ---- problem constants --------------------------------------------------
#define E_DIM 1024
#define HEADS 16
#define SEQ   2048
#define BATCH 2
#define HD    64
#define NTOK  (SEQ * BATCH)   // 4096 token rows, row index = t*B + b

typedef __attribute__((ext_vector_type(16))) __bf16 v16bf;
typedef __attribute__((ext_vector_type(8)))  __bf16 bf8v;
typedef __attribute__((ext_vector_type(4)))  __bf16 bf4v;
typedef __attribute__((ext_vector_type(4)))  float  f4v;
typedef __attribute__((ext_vector_type(8)))  float  v8f;

__device__ __forceinline__ bf8v ld8(const __bf16* p) {
  return *reinterpret_cast<const bf8v*>(p);
}
__device__ __forceinline__ v16bf cat16(bf8v lo, bf8v hi) {
  return __builtin_shufflevector(lo, hi, 0,1,2,3,4,5,6,7,8,9,10,11,12,13,14,15);
}
__device__ __forceinline__ v8f wmma_bf16(v16bf a, v16bf b, v8f c) {
  // D(16x16,f32) = A(16x32,bf16) x B(32x16,bf16) + C
  return __builtin_amdgcn_wmma_f32_16x16x32_bf16(false, a, false, b, (short)0, c,
                                                 false, false);
}

// ---- kernel 1: vectorized f32 -> bf16 conversion ------------------------
__global__ void cvt_f32_bf16(const float* __restrict__ s, __bf16* __restrict__ d, int n4) {
  int i = blockIdx.x * blockDim.x + threadIdx.x;
  int stride = gridDim.x * blockDim.x;
  const f4v* s4 = reinterpret_cast<const f4v*>(s);
  bf4v* d4 = reinterpret_cast<bf4v*>(d);
  for (; i < n4; i += stride) {
    f4v v = s4[i];
    bf4v o;
    o.x = (__bf16)v.x; o.y = (__bf16)v.y; o.z = (__bf16)v.z; o.w = (__bf16)v.w;
    d4[i] = o;
  }
}

// ---- kernel 2: fused QKV projection -------------------------------------
// One wave -> 32x16 output tile for all three matrices (6 WMMA / K-step).
// Ping-pong fragment buffers; unroll pinned to 1 so only two fragment sets
// are ever live (keeps allocation < 256 VGPRs, in-place WMMA accumulation).
struct QkvFrags { v16bf a0, a1, fq, fk, fv; };

__global__ void __launch_bounds__(128) qkv_gemm(
    const __bf16* __restrict__ xb,
    const __bf16* __restrict__ wq, const __bf16* __restrict__ wk,
    const __bf16* __restrict__ wv,
    const float* __restrict__ bq, const float* __restrict__ bk,
    const float* __restrict__ bv,
    __bf16* __restrict__ qO, __bf16* __restrict__ kO, __bf16* __restrict__ vT)
{
  const int lane = threadIdx.x & 31;
  const int wave = blockIdx.x * (blockDim.x >> 5) + (threadIdx.x >> 5);
  const int rowtile = wave >> 6;   // 0..127 (NTOK/32)
  const int coltile = wave & 63;   // 0..63  (E/16)
  const int l16 = lane & 15, half = lane >> 4;

  const __bf16* xr0 = xb + (size_t)(rowtile * 32 + l16) * E_DIM;
  const __bf16* xr1 = xr0 + (size_t)16 * E_DIM;
  const __bf16* wqr = wq + (size_t)(coltile * 16 + l16) * E_DIM;
  const __bf16* wkr = wk + (size_t)(coltile * 16 + l16) * E_DIM;
  const __bf16* wvr = wv + (size_t)(coltile * 16 + l16) * E_DIM;

  auto loadF = [&](int kk) {
    QkvFrags f;
    f.a0 = cat16(ld8(xr0 + kk + half * 8),  ld8(xr0 + kk + 16 + half * 8));
    f.a1 = cat16(ld8(xr1 + kk + half * 8),  ld8(xr1 + kk + 16 + half * 8));
    f.fq = cat16(ld8(wqr + kk + half * 8),  ld8(wqr + kk + 16 + half * 8));
    f.fk = cat16(ld8(wkr + kk + half * 8),  ld8(wkr + kk + 16 + half * 8));
    f.fv = cat16(ld8(wvr + kk + half * 8),  ld8(wvr + kk + 16 + half * 8));
    return f;
  };

  v8f aq0{}, aq1{}, ak0{}, ak1{}, av0{}, av1{};
  auto doWmma = [&](const QkvFrags& f) {
    aq0 = wmma_bf16(f.a0, f.fq, aq0);
    aq1 = wmma_bf16(f.a1, f.fq, aq1);
    ak0 = wmma_bf16(f.a0, f.fk, ak0);
    ak1 = wmma_bf16(f.a1, f.fk, ak1);
    av0 = wmma_bf16(f.a0, f.fv, av0);
    av1 = wmma_bf16(f.a1, f.fv, av1);
  };

  QkvFrags f0 = loadF(0);
  QkvFrags f1 = loadF(32);
#pragma unroll 1
  for (int kk = 64; kk < E_DIM; kk += 64) {
    doWmma(f0);
    f0 = loadF(kk);          // refill ping while pong computes
    doWmma(f1);
    f1 = loadF(kk + 32);
  }
  doWmma(f0);
  doWmma(f1);

  const float scaling = 0.125f;             // HD^-0.5
  const int j = coltile * 16 + l16;         // C/D layout: lane owns column j
  const int h = j >> 6, d = j & 63;
  const float bqv = bq[j], bkv = bk[j], bvv = bv[j];
#pragma unroll
  for (int rt = 0; rt < 2; ++rt) {
    const v8f& xq = rt ? aq1 : aq0;
    const v8f& xk = rt ? ak1 : ak0;
    const v8f& xv = rt ? av1 : av0;
#pragma unroll
    for (int r = 0; r < 8; ++r) {
      const int tok = rowtile * 32 + rt * 16 + half * 8 + r;
      const int t = tok >> 1, b = tok & 1;    // BATCH == 2
      const size_t qi = (((size_t)(b * HEADS + h)) * SEQ + t) * HD + d;
      qO[qi] = (__bf16)((xq[r] + bqv) * scaling);
      kO[qi] = (__bf16)(xk[r] + bkv);
      const size_t vi = (((size_t)(b * HEADS + h)) * HD + d) * SEQ + t;
      vT[vi] = (__bf16)(xv[r] + bvv);
    }
  }
}

// ---- kernel 3: flash attention ------------------------------------------
// One wave -> one (b,h) x 16-row query tile; streams keys in blocks of 32.
// Both 16-col S sub-tiles computed first, then ONE fused online-softmax
// update per block (single max-reduce / correction / sum-reduce).
__global__ void __launch_bounds__(128) attn_kernel(
    const __bf16* __restrict__ qB, const __bf16* __restrict__ kB,
    const __bf16* __restrict__ vT,
    const float* __restrict__ bias, const int* __restrict__ pad,
    __bf16* __restrict__ attnB)
{
  __shared__ __bf16 Pl[4][16 * 32];         // per-wave P transpose buffer (1KB)
  const int lane = threadIdx.x & 31;
  const int wslot = threadIdx.x >> 5;
  const int wave = blockIdx.x * 4 + wslot;
  const int bh = wave >> 7;                 // 0..31
  const int ttile = wave & 127;             // 0..127
  const int l16 = lane & 15, half = lane >> 4;
  const int b = bh >> 4, h = bh & (HEADS - 1);

  // Q A-fragments (16 rows x K=64) held in registers for the whole scan
  const __bf16* qr = qB + ((size_t)bh * SEQ + ttile * 16 + l16) * HD;
  const v16bf qa0 = cat16(ld8(qr + half * 8),      ld8(qr + 16 + half * 8));
  const v16bf qa1 = cat16(ld8(qr + 32 + half * 8), ld8(qr + 48 + half * 8));

  float m_run[8], l_run[8];
#pragma unroll
  for (int r = 0; r < 8; ++r) { m_run[r] = -1e30f; l_run[r] = 0.0f; }
  v8f acc[4] = {};

  const float* biasRow = bias + ((size_t)bh * SEQ + ttile * 16) * SEQ;
  const int*   padRow  = pad + b * SEQ;
  __bf16*      Pw      = &Pl[wslot][0];

#pragma unroll 1
  for (int sb = 0; sb < SEQ; sb += 32) {
    // V fragments for this block, issued early so S-WMMA + softmax hide them
    v16bf vf[4];
#pragma unroll
    for (int c = 0; c < 4; ++c) {
      const __bf16* vr = vT + ((size_t)bh * HD + c * 16 + l16) * SEQ + sb;
      vf[c] = cat16(ld8(vr + half * 8), ld8(vr + 16 + half * 8));
    }
    // prefetch next block's bias stream (HBM-dominant) into L2
    if (sb + 32 < SEQ) {
#pragma unroll
      for (int r = 0; r < 8; ++r)
        __builtin_prefetch(biasRow + (size_t)(half * 8 + r) * SEQ + sb + 32 + l16, 0, 0);
    }

    // ---- both S sub-tiles (16x16 each) ----
    const __bf16* kr0 = kB + ((size_t)bh * SEQ + sb + l16) * HD;
    const __bf16* kr1 = kr0 + (size_t)16 * HD;
    const v16bf kf00 = cat16(ld8(kr0 + half * 8),      ld8(kr0 + 16 + half * 8));
    const v16bf kf01 = cat16(ld8(kr0 + 32 + half * 8), ld8(kr0 + 48 + half * 8));
    const v16bf kf10 = cat16(ld8(kr1 + half * 8),      ld8(kr1 + 16 + half * 8));
    const v16bf kf11 = cat16(ld8(kr1 + 32 + half * 8), ld8(kr1 + 48 + half * 8));
    v8f S0{}, S1{};
    S0 = wmma_bf16(qa0, kf00, S0);
    S0 = wmma_bf16(qa1, kf01, S0);
    S1 = wmma_bf16(qa0, kf10, S1);
    S1 = wmma_bf16(qa1, kf11, S1);

    const int scol0 = sb + l16, scol1 = scol0 + 16;
    const bool masked0 = padRow[scol0] != 0;
    const bool masked1 = padRow[scol1] != 0;
    const float* bp = biasRow + (size_t)(half * 8) * SEQ;
    float w0[8], w1[8];
#pragma unroll
    for (int r = 0; r < 8; ++r) {
      float v0 = masked0 ? -1e-16f : S0[r];            // where(pad, -1e-16, qk)
      float v1 = masked1 ? -1e-16f : S1[r];
      v0 += __builtin_nontemporal_load(bp + (size_t)r * SEQ + scol0);
      v1 += __builtin_nontemporal_load(bp + (size_t)r * SEQ + scol1);
      w0[r] = v0; w1[r] = v1;
    }

    // ---- fused online softmax over the full 32-key block ----
#pragma unroll
    for (int r = 0; r < 8; ++r) {
      float mx = fmaxf(w0[r], w1[r]);
#pragma unroll
      for (int off = 1; off < 16; off <<= 1)
        mx = fmaxf(mx, __shfl_xor(mx, off, 32));
      const float newm = fmaxf(m_run[r], mx);
      const float corr = __expf(m_run[r] - newm);
      const float p0 = __expf(w0[r] - newm);
      const float p1 = __expf(w1[r] - newm);
      float ps = p0 + p1;
#pragma unroll
      for (int off = 1; off < 16; off <<= 1)
        ps += __shfl_xor(ps, off, 32);
      l_run[r] = l_run[r] * corr + ps;
      m_run[r] = newm;
#pragma unroll
      for (int c = 0; c < 4; ++c) acc[c][r] *= corr;
      Pw[(half * 8 + r) * 32 + l16]      = (__bf16)p0;  // C-layout -> LDS
      Pw[(half * 8 + r) * 32 + 16 + l16] = (__bf16)p1;
    }

    asm volatile("s_wait_dscnt 0x0" ::: "memory");   // wave-private LDS transpose
    // P back as A-fragment (16 x 32)
    const __bf16* pr = Pw + l16 * 32;
    const v16bf pa = cat16(ld8(pr + half * 8), ld8(pr + 16 + half * 8));
#pragma unroll
    for (int c = 0; c < 4; ++c)
      acc[c] = wmma_bf16(pa, vf[c], acc[c]);
  }

  // normalize + store attn in (T,B,E) token-major bf16 for the output GEMM
#pragma unroll
  for (int c = 0; c < 4; ++c) {
    const int dcol = c * 16 + l16;
#pragma unroll
    for (int r = 0; r < 8; ++r) {
      const int t = ttile * 16 + half * 8 + r;
      const float o = acc[c][r] / l_run[r];
      attnB[((size_t)t * BATCH + b) * E_DIM + h * HD + dcol] = (__bf16)o;
    }
  }
}

// ---- kernel 4: output projection (32x32 tile / wave, ping-pong) ----------
struct OutFrags { v16bf a0, a1, f0, f1; };

__global__ void __launch_bounds__(128) out_gemm(
    const __bf16* __restrict__ attnB, const __bf16* __restrict__ wo,
    const float* __restrict__ bo, float* __restrict__ out)
{
  const int lane = threadIdx.x & 31;
  const int wave = blockIdx.x * (blockDim.x >> 5) + (threadIdx.x >> 5);
  const int rowtile = wave >> 5;   // 0..127 (NTOK/32)
  const int coltile = wave & 31;   // 0..31  (E/32)
  const int l16 = lane & 15, half = lane >> 4;

  const __bf16* ar0 = attnB + (size_t)(rowtile * 32 + l16) * E_DIM;
  const __bf16* ar1 = ar0 + (size_t)16 * E_DIM;
  const __bf16* wr0 = wo + (size_t)(coltile * 32 + l16) * E_DIM;
  const __bf16* wr1 = wr0 + (size_t)16 * E_DIM;

  auto loadF = [&](int kk) {
    OutFrags f;
    f.a0 = cat16(ld8(ar0 + kk + half * 8),  ld8(ar0 + kk + 16 + half * 8));
    f.a1 = cat16(ld8(ar1 + kk + half * 8),  ld8(ar1 + kk + 16 + half * 8));
    f.f0 = cat16(ld8(wr0 + kk + half * 8),  ld8(wr0 + kk + 16 + half * 8));
    f.f1 = cat16(ld8(wr1 + kk + half * 8),  ld8(wr1 + kk + 16 + half * 8));
    return f;
  };

  v8f a00{}, a01{}, a10{}, a11{};
  auto doWmma = [&](const OutFrags& f) {
    a00 = wmma_bf16(f.a0, f.f0, a00);
    a01 = wmma_bf16(f.a0, f.f1, a01);
    a10 = wmma_bf16(f.a1, f.f0, a10);
    a11 = wmma_bf16(f.a1, f.f1, a11);
  };

  OutFrags f0 = loadF(0);
  OutFrags f1 = loadF(32);
#pragma unroll 1
  for (int kk = 64; kk < E_DIM; kk += 64) {
    doWmma(f0);
    f0 = loadF(kk);
    doWmma(f1);
    f1 = loadF(kk + 32);
  }
  doWmma(f0);
  doWmma(f1);

#pragma unroll
  for (int jc = 0; jc < 2; ++jc) {
    const int j = coltile * 32 + jc * 16 + l16;
    const float bov = bo[j];
#pragma unroll
    for (int rt = 0; rt < 2; ++rt) {
      const v8f& av = rt ? (jc ? a11 : a10) : (jc ? a01 : a00);
#pragma unroll
      for (int r = 0; r < 8; ++r) {
        const int tok = rowtile * 32 + rt * 16 + half * 8 + r;
        out[(size_t)tok * E_DIM + j] = av[r] + bov;
      }
    }
  }
}

// ---- launcher ------------------------------------------------------------
extern "C" void kernel_launch(void* const* d_in, const int* in_sizes, int n_in,
                              void* d_out, int out_size, void* d_ws, size_t ws_size,
                              hipStream_t stream) {
  (void)in_sizes; (void)n_in; (void)out_size; (void)ws_size;
  const float* x    = (const float*)d_in[0];
  const int*   pad  = (const int*)d_in[1];
  const float* bias = (const float*)d_in[2];
  const float* Wq   = (const float*)d_in[3];
  const float* bq   = (const float*)d_in[4];
  const float* Wk   = (const float*)d_in[5];
  const float* bk   = (const float*)d_in[6];
  const float* Wv   = (const float*)d_in[7];
  const float* bv   = (const float*)d_in[8];
  const float* Wo   = (const float*)d_in[9];
  const float* bo   = (const float*)d_in[10];
  float* out = (float*)d_out;

  char* ws = (char*)d_ws;
  const size_t MB = 1u << 20;
  __bf16* xb    = (__bf16*)(ws +  0 * MB);   // 4096x1024        (8 MB)
  __bf16* wqb   = (__bf16*)(ws +  8 * MB);   // 1024x1024        (2 MB)
  __bf16* wkb   = (__bf16*)(ws + 10 * MB);
  __bf16* wvb   = (__bf16*)(ws + 12 * MB);
  __bf16* wob   = (__bf16*)(ws + 14 * MB);
  __bf16* qb    = (__bf16*)(ws + 16 * MB);   // [B,H,T,64]       (8 MB)
  __bf16* kb    = (__bf16*)(ws + 24 * MB);   // [B,H,T,64]
  __bf16* vtb   = (__bf16*)(ws + 32 * MB);   // [B,H,64,T]
  __bf16* attnb = (__bf16*)(ws + 40 * MB);   // 4096x1024

  cvt_f32_bf16<<<1024, 256, 0, stream>>>(x,  xb,  (NTOK * E_DIM) / 4);
  cvt_f32_bf16<<<256, 256, 0, stream>>>(Wq, wqb, (E_DIM * E_DIM) / 4);
  cvt_f32_bf16<<<256, 256, 0, stream>>>(Wk, wkb, (E_DIM * E_DIM) / 4);
  cvt_f32_bf16<<<256, 256, 0, stream>>>(Wv, wvb, (E_DIM * E_DIM) / 4);
  cvt_f32_bf16<<<256, 256, 0, stream>>>(Wo, wob, (E_DIM * E_DIM) / 4);

  qkv_gemm<<<2048, 128, 0, stream>>>(xb, wqb, wkb, wvb, bq, bk, bv, qb, kb, vtb);
  attn_kernel<<<1024, 128, 0, stream>>>(qb, kb, vtb, bias, pad, attnb);
  out_gemm<<<1024, 128, 0, stream>>>(attnb, wob, bo, out);
}
